// MyTrans_44607530336841
// MI455X (gfx1250) — matrix operand order repaired and checked by reading, not compiled, not run on hardware
//
#include <hip/hip_runtime.h>
#include <hip/hip_bf16.h>
#include <math.h>

typedef __attribute__((ext_vector_type(16))) _Float16 v16h;
typedef __attribute__((ext_vector_type(8)))  float    v8f;

#define TPB   256
#define WAVES 8

// ---------------- f16 weight-workspace layout (element offsets) ----------------
enum : int {
  OFF_A1WQ = 0,      OFF_A1WK = 4096,   OFF_A1WV = 8192,   OFF_A1WO = 12288,
  OFF_A2WQ = 28672,  OFF_A2WK = 32768,  OFF_A2WV = 36864,  OFF_A2WO = 40960,
  OFF_CAM  = 57344,                      // 25 x 128 x 128 (per-shift packed)
  OFF_PAM  = 466944,                     // 25 x 128 x 128
  OFF_PAMWV = 876544, OFF_PAMWQ = 892928, OFF_PAMWK = 894976,
  OFF_MLP1 = 897024,  OFF_MLP2 = 913408, WS_TOTAL = 929792
};

// ---------------- WMMA fragment helpers (ISA 7.12.2 layouts) ----------------
__device__ __forceinline__ v8f zeroC() {
  v8f z;
#pragma unroll
  for (int e = 0; e < 8; ++e) z[e] = 0.0f;
  return z;
}
__device__ __forceinline__ v16h zeroA() {
  v16h z;
#pragma unroll
  for (int i = 0; i < 16; ++i) z[i] = (_Float16)0.0f;
  return z;
}
// A fragment 16x32 f16, row-major source, stride ld (f16 elems).
// lanes 0-15 hold k 0..7 & 16..23 ; lanes 16-31 hold k 8..15 & 24..31.
__device__ __forceinline__ v16h frag_a(const _Float16* base, int r0, int k0, int ld, int lane) {
  const _Float16* p = base + (r0 + (lane & 15)) * ld + k0 + ((lane & 16) ? 8 : 0);
  v16h a;
  *reinterpret_cast<float4*>(&a)       = *reinterpret_cast<const float4*>(p);
  *(reinterpret_cast<float4*>(&a) + 1) = *reinterpret_cast<const float4*>(p + 16);
  return a;
}
// A fragment with spatial shift + zero padding (rows are 7x7 positions, r>=49 pad)
__device__ __forceinline__ v16h frag_a_shift(const _Float16* xi, int r0, int k0, int ld,
                                             int dy, int dx, int lane) {
  const int r = r0 + (lane & 15);
  const int y = r / 7 + dy, xx = r % 7 + dx;
  const bool valid = (r < 49) && ((unsigned)y < 7u) && ((unsigned)xx < 7u);
  v16h a = zeroA();
  if (valid) {
    const _Float16* p = xi + (y * 7 + xx) * ld + k0 + ((lane & 16) ? 8 : 0);
    *reinterpret_cast<float4*>(&a)       = *reinterpret_cast<const float4*>(p);
    *(reinterpret_cast<float4*>(&a) + 1) = *reinterpret_cast<const float4*>(p + 16);
  }
  return a;
}
// B fragment from Bt stored row-major (N,K): lanes 0-15 k 0..7 & 16..23, lanes 16-31 k 8..15 & 24..31.
__device__ __forceinline__ v16h frag_bt(const _Float16* bt, int c0, int k0, int ldb, int lane) {
  const _Float16* p = bt + (c0 + (lane & 15)) * ldb + k0 + ((lane & 16) ? 8 : 0);
  v16h b;
  *reinterpret_cast<float4*>(&b)       = *reinterpret_cast<const float4*>(p);
  *(reinterpret_cast<float4*>(&b) + 1) = *reinterpret_cast<const float4*>(p + 16);
  return b;
}
__device__ __forceinline__ v8f frag_loadc(const float* C, int r0, int c0, int ld, int lane) {
  const int col = c0 + (lane & 15), rb = r0 + ((lane & 16) ? 8 : 0);
  v8f v;
#pragma unroll
  for (int e = 0; e < 8; ++e) v[e] = C[(rb + e) * ld + col];
  return v;
}
__device__ __forceinline__ void frag_storec(float* C, v8f a, int r0, int c0, int ld, int lane) {
  const int col = c0 + (lane & 15), rb = r0 + ((lane & 16) ? 8 : 0);
#pragma unroll
  for (int e = 0; e < 8; ++e) C[(rb + e) * ld + col] = a[e];
}
__device__ __forceinline__ v8f wmma16(v16h a, v16h b, v8f c) {
  return __builtin_amdgcn_wmma_f32_16x16x32_f16(false, a, false, b, (short)0, c, false, false);
}

// ---------------- generic tiled GEMMs over LDS/global f16 ----------------
// D(M,N) f32 = A(M,K) * Bt(N,K)^T  (+C if accum). M,N mult of 16, K mult of 32.
// 2-wide N blocking (shared A fragment, 2 independent WMMA chains) when nt even.
__device__ __forceinline__ void mm_f32(const _Float16* A, const _Float16* Bt, float* C,
                                       int M, int N, int K, int lda, int ldb, int ldc,
                                       bool accum, int tid) {
  const int lane = tid & 31, wave = tid >> 5;
  const int mt = M >> 4, nt = N >> 4;
  if ((nt & 1) == 0) {
    const int np = nt >> 1, tiles = mt * np;
    for (int t = wave; t < tiles; t += WAVES) {
      const int r0 = (t / np) << 4, c0 = (t % np) << 5;
      v8f a0 = accum ? frag_loadc(C, r0, c0, ldc, lane) : zeroC();
      v8f a1 = accum ? frag_loadc(C, r0, c0 + 16, ldc, lane) : zeroC();
      for (int k = 0; k < K; k += 32) {
        const v16h af = frag_a(A, r0, k, lda, lane);
        a0 = wmma16(af, frag_bt(Bt, c0, k, ldb, lane), a0);
        a1 = wmma16(af, frag_bt(Bt, c0 + 16, k, ldb, lane), a1);
      }
      frag_storec(C, a0, r0, c0, ldc, lane);
      frag_storec(C, a1, r0, c0 + 16, ldc, lane);
    }
  } else {
    const int tiles = mt * nt;
    for (int t = wave; t < tiles; t += WAVES) {
      const int r0 = (t / nt) << 4, c0 = (t % nt) << 4;
      v8f acc = accum ? frag_loadc(C, r0, c0, ldc, lane) : zeroC();
      for (int k = 0; k < K; k += 32)
        acc = wmma16(frag_a(A, r0, k, lda, lane), frag_bt(Bt, c0, k, ldb, lane), acc);
      frag_storec(C, acc, r0, c0, ldc, lane);
    }
  }
}
// f16 output variant with optional per-row bias (nt assumed even: all call sites)
__device__ __forceinline__ void mm_f16h(const _Float16* A, const _Float16* Bt, _Float16* Ch,
                                        int M, int N, int K, int lda, int ldb, int ldc,
                                        const float* rowBias, int tid) {
  const int lane = tid & 31, wave = tid >> 5;
  const int mt = M >> 4, np = N >> 5, tiles = mt * np;
  for (int t = wave; t < tiles; t += WAVES) {
    const int r0 = (t / np) << 4, c0 = (t % np) << 5;
    v8f a0 = zeroC(), a1 = zeroC();
    for (int k = 0; k < K; k += 32) {
      const v16h af = frag_a(A, r0, k, lda, lane);
      a0 = wmma16(af, frag_bt(Bt, c0, k, ldb, lane), a0);
      a1 = wmma16(af, frag_bt(Bt, c0 + 16, k, ldb, lane), a1);
    }
    const int rb = r0 + ((lane & 16) ? 8 : 0);
#pragma unroll
    for (int e = 0; e < 8; ++e) {
      float v0 = a0[e], v1 = a1[e];
      if (rowBias) { v0 += rowBias[rb + e]; v1 += rowBias[rb + e]; }
      Ch[(rb + e) * ldc + c0 + (lane & 15)]      = (_Float16)v0;
      Ch[(rb + e) * ldc + c0 + 16 + (lane & 15)] = (_Float16)v1;
    }
  }
}
// f16 output with per-col bias + exact GELU epilogue (nt even)
__device__ __forceinline__ void mm_gelu(const _Float16* A, const _Float16* Bt, _Float16* Ch,
                                        int M, int N, int K, int lda, int ldb, int ldc,
                                        const float* colBias, int tid) {
  const int lane = tid & 31, wave = tid >> 5;
  const int mt = M >> 4, np = N >> 5, tiles = mt * np;
  for (int t = wave; t < tiles; t += WAVES) {
    const int r0 = (t / np) << 4, c0 = (t % np) << 5;
    v8f a0 = zeroC(), a1 = zeroC();
    for (int k = 0; k < K; k += 32) {
      const v16h af = frag_a(A, r0, k, lda, lane);
      a0 = wmma16(af, frag_bt(Bt, c0, k, ldb, lane), a0);
      a1 = wmma16(af, frag_bt(Bt, c0 + 16, k, ldb, lane), a1);
    }
    const int col0 = c0 + (lane & 15), rb = r0 + ((lane & 16) ? 8 : 0);
    const float b0 = colBias[col0], b1 = colBias[col0 + 16];
#pragma unroll
    for (int e = 0; e < 8; ++e) {
      float v0 = a0[e] + b0, v1 = a1[e] + b1;
      v0 = 0.5f * v0 * (1.0f + erff(v0 * 0.70710678118654752f));
      v1 = 0.5f * v1 * (1.0f + erff(v1 * 0.70710678118654752f));
      Ch[(rb + e) * ldc + col0]      = (_Float16)v0;
      Ch[(rb + e) * ldc + col0 + 16] = (_Float16)v1;
    }
  }
}

// 5x5 conv(128->128) on 7x7 as 25 shifted GEMMs.
// Each wave: one 16-row stripe x four 16-col tiles -> 4 independent WMMA chains,
// register-resident accumulators across all 100 k-steps, shared A fragment.
// epilogue: out[n,c] (= or +=) gamma*pre + xi + relu(bn(conv + bias))
__device__ __forceinline__ void conv_block(const _Float16* Ah, const _Float16* Wp,
                                           const float* bc, const float* bnw, const float* bnb,
                                           float gamma, const float* pre, bool preCMajor,
                                           const float* xi, float* outp, bool addOut, int tid) {
  const int lane = tid & 31, wave = tid >> 5;
  for (int i = tid; i < 25 * 128; i += TPB)                 // pull weight slices toward L2/L0
    __builtin_prefetch(Wp + (size_t)i * 128, 0, 0);
  const float bns = rsqrtf(1.0f + 1e-5f);
  const int r0 = (wave & 3) << 4;          // row stripe 0..3
  const int c0 = (wave >> 2) << 6;         // col group 0 or 64
  v8f accs[4];
#pragma unroll
  for (int i = 0; i < 4; ++i) accs[i] = zeroC();
  for (int s = 0; s < 25; ++s) {
    const int dy = s / 5 - 2, dx = s % 5 - 2;
    const _Float16* w = Wp + s * (128 * 128);
#pragma unroll
    for (int k = 0; k < 128; k += 32) {
      const v16h af = frag_a_shift(Ah, r0, k, 128, dy, dx, lane);
#pragma unroll
      for (int i = 0; i < 4; ++i)
        accs[i] = wmma16(af, frag_bt(w, c0 + 16 * i, k, 128, lane), accs[i]);
    }
  }
#pragma unroll
  for (int i = 0; i < 4; ++i) {
    const int col = c0 + 16 * i + (lane & 15), rb = r0 + ((lane & 16) ? 8 : 0);
    const float sc = bnw[col] * bns, bb = bnb[col], bcv = bc[col];
#pragma unroll
    for (int e = 0; e < 8; ++e) {
      const int n = rb + e;
      const float rl = fmaxf((accs[i][e] + bcv) * sc + bb, 0.0f);
      const float pv = preCMajor ? pre[col * 64 + n] : pre[n * 128 + col];
      const float val = gamma * pv + xi[n * 128 + col] + rl;
      if (addOut) outp[n * 128 + col] += val;
      else        outp[n * 128 + col]  = val;
    }
  }
}

// ---------------- AGA block ----------------
__device__ __forceinline__ void aga_block(const _Float16* xA, const _Float16* xB,
                                          const _Float16* wq, const _Float16* wk,
                                          const _Float16* wv, const _Float16* wo,
                                          _Float16* qh, _Float16* kh, _Float16* vh,
                                          _Float16* o0h, _Float16* cath, float* xa,
                                          bool accum, int tid) {
  mm_f16h(xA, wq, qh, 64, 64, 64, 64, 64, 64, nullptr, tid);
  mm_f16h(xB, wk, kh, 64, 64, 64, 64, 64, 64, nullptr, tid);
  mm_f16h(xB, wv, vh, 64, 64, 64, 64, 64, 64, nullptr, tid);
  __syncthreads();
  const float scale = 0.35355339059327373f;
  // out0 = softmax(q k^T * s) v    (per (head,row); H=8, Dh=8, N=49)
  for (int r = tid; r < 8 * 49; r += TPB) {
    const int hd = r / 49, i = r - hd * 49, off = hd * 8;
    float qv[8];
#pragma unroll
    for (int d = 0; d < 8; ++d) qv[d] = (float)qh[i * 64 + off + d];
    float mx = -1e30f;
    for (int j = 0; j < 49; ++j) {
      float s = 0.f;
#pragma unroll
      for (int d = 0; d < 8; ++d) s += qv[d] * (float)kh[j * 64 + off + d];
      mx = fmaxf(mx, s * scale);
    }
    float den = 0.f, num[8] = {0, 0, 0, 0, 0, 0, 0, 0};
    for (int j = 0; j < 49; ++j) {
      float s = 0.f;
#pragma unroll
      for (int d = 0; d < 8; ++d) s += qv[d] * (float)kh[j * 64 + off + d];
      const float p = __expf(s * scale - mx);
      den += p;
#pragma unroll
      for (int d = 0; d < 8; ++d) num[d] += p * (float)vh[j * 64 + off + d];
    }
    const float inv = 1.f / den;
#pragma unroll
    for (int d = 0; d < 8; ++d) o0h[i * 64 + off + d] = (_Float16)(num[d] * inv);
  }
  __syncthreads();
  // out1 = softmax(q q^T * s) out0 -> cat[:, :64]
  for (int r = tid; r < 8 * 49; r += TPB) {
    const int hd = r / 49, i = r - hd * 49, off = hd * 8;
    float qv[8];
#pragma unroll
    for (int d = 0; d < 8; ++d) qv[d] = (float)qh[i * 64 + off + d];
    float mx = -1e30f;
    for (int j = 0; j < 49; ++j) {
      float s = 0.f;
#pragma unroll
      for (int d = 0; d < 8; ++d) s += qv[d] * (float)qh[j * 64 + off + d];
      mx = fmaxf(mx, s * scale);
    }
    float den = 0.f, num[8] = {0, 0, 0, 0, 0, 0, 0, 0};
    for (int j = 0; j < 49; ++j) {
      float s = 0.f;
#pragma unroll
      for (int d = 0; d < 8; ++d) s += qv[d] * (float)qh[j * 64 + off + d];
      const float p = __expf(s * scale - mx);
      den += p;
#pragma unroll
      for (int d = 0; d < 8; ++d) num[d] += p * (float)o0h[j * 64 + off + d];
    }
    const float inv = 1.f / den;
#pragma unroll
    for (int d = 0; d < 8; ++d) cath[i * 128 + off + d] = (_Float16)(num[d] * inv);
  }
  // cat[:, 64:] = xA ; zero pad rows
  for (int idx = tid; idx < 64 * 128; idx += TPB) {
    const int n = idx >> 7, c = idx & 127;
    if (c >= 64)      cath[idx] = (n < 49) ? xA[n * 64 + (c - 64)] : (_Float16)0.f;
    else if (n >= 49) cath[idx] = (_Float16)0.f;
  }
  __syncthreads();
  mm_f32(cath, wo, xa, 64, 128, 128, 128, 128, 128, accum, tid);
}

__device__ __forceinline__ void cam_softmax_row(const float* e, _Float16* o) {
  float mn = 1e30f;
  for (int d = 0; d < 128; ++d) mn = fminf(mn, e[d]);       // softmax(rowmax - e): max cancels
  float den = 0.f;
  for (int d = 0; d < 128; ++d) den += __expf(mn - e[d]);
  const float inv = 1.f / den;
  for (int d = 0; d < 128; ++d) o[d] = (_Float16)(__expf(mn - e[d]) * inv);
}

// ---------------- weight conversion / packing kernel ----------------
__global__ void __launch_bounds__(TPB) prep_weights(
    const float* __restrict__ a1wq, const float* __restrict__ a1wk,
    const float* __restrict__ a1wv, const float* __restrict__ a1wo,
    const float* __restrict__ a2wq, const float* __restrict__ a2wk,
    const float* __restrict__ a2wv, const float* __restrict__ a2wo,
    const float* __restrict__ camwc, const float* __restrict__ pamwc,
    const float* __restrict__ pamwv, const float* __restrict__ pamwq,
    const float* __restrict__ pamwk, const float* __restrict__ mlpw1,
    const float* __restrict__ mlpw2, _Float16* __restrict__ ws) {
  const int i = blockIdx.x * TPB + threadIdx.x;
  if (i >= WS_TOTAL) return;
  float v;
  if (i < OFF_A1WO)
    v = (i < OFF_A1WK) ? a1wq[i] : (i < OFF_A1WV) ? a1wk[i - OFF_A1WK] : a1wv[i - OFF_A1WV];
  else if (i < OFF_A2WQ) v = a1wo[i - OFF_A1WO];
  else if (i < OFF_A2WO)
    v = (i < OFF_A2WK) ? a2wq[i - OFF_A2WQ]
                       : (i < OFF_A2WV) ? a2wk[i - OFF_A2WK] : a2wv[i - OFF_A2WV];
  else if (i < OFF_CAM) v = a2wo[i - OFF_A2WO];
  else if (i < OFF_PAM) {                                   // pack w(o,c,s) -> [s][o][c]
    const int j = i - OFF_CAM, s = j >> 14, r = j & 16383;
    v = camwc[(r >> 7) * 3200 + (r & 127) * 25 + s];
  } else if (i < OFF_PAMWV) {
    const int j = i - OFF_PAM, s = j >> 14, r = j & 16383;
    v = pamwc[(r >> 7) * 3200 + (r & 127) * 25 + s];
  } else if (i < OFF_PAMWQ) v = pamwv[i - OFF_PAMWV];
  else if (i < OFF_PAMWK)   v = pamwq[i - OFF_PAMWQ];
  else if (i < OFF_MLP1)    v = pamwk[i - OFF_PAMWK];
  else if (i < OFF_MLP2)    v = mlpw1[i - OFF_MLP1];
  else                      v = mlpw2[i - OFF_MLP2];
  ws[i] = (_Float16)v;
}

// ---------------- fused per-sample network: 1 workgroup = 1 sample ----------------
__global__ void __launch_bounds__(TPB) fused_net(
    const float* __restrict__ x,
    const float* __restrict__ w3d, const float* __restrict__ b3d,
    const float* __restrict__ bn3dw, const float* __restrict__ bn3db,
    const float* __restrict__ w2d, const float* __restrict__ b2d,
    const float* __restrict__ bn2dw, const float* __restrict__ bn2db,
    const float* __restrict__ a1bo, const float* __restrict__ a2bo,
    const float* __restrict__ camg_, const float* __restrict__ cambc,
    const float* __restrict__ cambnw, const float* __restrict__ cambnb,
    const float* __restrict__ pambq, const float* __restrict__ pambk,
    const float* __restrict__ pambv, const float* __restrict__ pamg_,
    const float* __restrict__ pambc, const float* __restrict__ pambnw,
    const float* __restrict__ pambnb,
    const float* __restrict__ mlpb1, const float* __restrict__ mlpb2,
    const float* __restrict__ fcw, const float* __restrict__ fcb,
    const float* __restrict__ lnw, const float* __restrict__ lnb,
    const _Float16* __restrict__ ws, float* __restrict__ out) {
  extern __shared__ char smem[];                            // 160 KB dynamic LDS
  _Float16* R1h    = (_Float16*)(smem + 0);                 // 16KB: x1h|x2h ; oc'_h ; op'_h ; xm_h
  _Float16* x1h    = R1h;                                   // 64x64
  _Float16* x2h    = R1h + 64 * 64;
  _Float16* qh     = (_Float16*)(smem + 16384);             // 24KB region
  _Float16* kh     = qh + 4096;
  _Float16* vh     = qh + 8192;
  _Float16* pattnh = qh;                                    // 64x64 (phase E)
  _Float16* pvh    = qh + 4096;                             // 128x64 (phase E)
  _Float16* hh     = qh;                                    // 64x128 (phase F)
  _Float16* o0h    = (_Float16*)(smem + 40960);             // 8KB: out0 ; later pq/pk f32
  float*    R3f    = (float*)(smem + 40960);
  _Float16* R4h    = (_Float16*)(smem + 49152);             // 16KB: cat ; xich ; xih
  float*    R5f    = (float*)(smem + 65536);                // 32KB: xa / xi  f32 (n-major)
  float*    R6f    = (float*)(smem + 98304);                // 32KB: s_in|w2s|w3s ; energy ; oc ; op ; h2
  float*    s_in   = R6f;                                   // 14*49 (phase A only)
  float*    w2s    = R6f + 704;                             // 6400  (phase A only)
  float*    w3s    = R6f + 7104;                            // 640   (phase A only)
  _Float16* R7h    = (_Float16*)(smem + 131072);            // 32KB: cattn f16 128x128
  float*    R7f    = (float*)(smem + 131072);               //        x_cam / xm f32 64x128

  const int tid = threadIdx.x;
  const int b   = blockIdx.x;
  const float bns = rsqrtf(1.0f + 1e-5f);

  // ---- Phase A: load input + small weights, 1x1 conv(10->64), 5x5 conv(4->64), BN+ReLU ----
  const float* xp = x + (size_t)b * (14 * 49);
  for (int i = tid; i < 14 * 49; i += TPB) s_in[i] = xp[i];
  for (int i = tid; i < 6400; i += TPB) w2s[i] = w2d[i];
  for (int i = tid; i < 640; i += TPB) w3s[i] = w3d[i];
  __syncthreads();
  for (int idx = tid; idx < 64 * 64; idx += TPB) {
    const int n = idx >> 6, o = idx & 63;
    _Float16 r1 = (_Float16)0.f, r2 = (_Float16)0.f;
    if (n < 49) {
      float a1 = b3d[o];
      for (int c = 0; c < 10; ++c) a1 += s_in[c * 49 + n] * w3s[o * 10 + c];
      r1 = (_Float16)fmaxf(a1 * (bn3dw[o] * bns) + bn3db[o], 0.f);
      const int y = n / 7, xc = n % 7;
      float a2 = b2d[o];
      for (int c = 0; c < 4; ++c)
        for (int ky = 0; ky < 5; ++ky) {
          const int yy = y + ky - 2;
          if ((unsigned)yy >= 7u) continue;
          for (int kx = 0; kx < 5; ++kx) {
            const int xx = xc + kx - 2;
            if ((unsigned)xx >= 7u) continue;
            a2 += s_in[(10 + c) * 49 + yy * 7 + xx] * w2s[o * 100 + c * 25 + ky * 5 + kx];
          }
        }
      r2 = (_Float16)fmaxf(a2 * (bn2dw[o] * bns) + bn2db[o], 0.f);
    }
    x1h[idx] = r1;
    x2h[idx] = r2;
  }
  __syncthreads();

  // ---- Phase B: xa = AGA(x1,x2) + AGA(x2,x1) -> R5f (xi) ----
  aga_block(x1h, x2h, ws + OFF_A1WQ, ws + OFF_A1WK, ws + OFF_A1WV, ws + OFF_A1WO,
            qh, kh, vh, o0h, R4h, R5f, false, tid);
  __syncthreads();
  aga_block(x2h, x1h, ws + OFF_A2WQ, ws + OFF_A2WK, ws + OFF_A2WV, ws + OFF_A2WO,
            qh, kh, vh, o0h, R4h, R5f, true, tid);
  __syncthreads();
  for (int idx = tid; idx < 49 * 128; idx += TPB) {
    const int c = idx & 127;
    R5f[idx] += a1bo[c] + a2bo[c];
  }
  __syncthreads();

  // ---- Phase D: CAM ----
  // xich (c-major f16) for energy
  for (int idx = tid; idx < 128 * 64; idx += TPB) {
    const int c = idx >> 6, n = idx & 63;
    R4h[idx] = (n < 49) ? (_Float16)R5f[n * 128 + c] : (_Float16)0.f;
  }
  __syncthreads();
  mm_f32(R4h, R4h, R6f, 64, 128, 64, 64, 64, 128, false, tid);   // energy rows 0..63
  __syncthreads();
  for (int c = tid; c < 64; c += TPB) cam_softmax_row(R6f + c * 128, R7h + c * 128);
  __syncthreads();
  mm_f32(R4h + 64 * 64, R4h, R6f, 64, 128, 64, 64, 64, 128, false, tid); // rows 64..127
  __syncthreads();
  for (int c = tid; c < 64; c += TPB) cam_softmax_row(R6f + c * 128, R7h + (64 + c) * 128);
  __syncthreads();
  // xih (n-major f16): reused for oc, pv, pq/pk
  for (int idx = tid; idx < 64 * 128; idx += TPB) {
    const int n = idx >> 7;
    R4h[idx] = (n < 49) ? (_Float16)R5f[idx] : (_Float16)0.f;
  }
  __syncthreads();
  mm_f32(R7h, R4h, R6f, 128, 64, 128, 128, 128, 64, false, tid); // oc (c-major) = cattn @ f
  __syncthreads();
  const float camg = camg_[0];
  for (int idx = tid; idx < 64 * 128; idx += TPB) {              // oc' f16 (n-major) -> R1
    const int n = idx >> 7, c = idx & 127;
    R1h[idx] = (n < 49) ? (_Float16)(camg * R6f[c * 64 + n] + R5f[idx]) : (_Float16)0.f;
  }
  __syncthreads();
  conv_block(R1h, ws + OFF_CAM, cambc, cambnw, cambnb, camg, R6f, true, R5f, R7f, false, tid);
  __syncthreads();

  // ---- Phase E: PAM ----
  mm_f32(R4h, ws + OFF_PAMWQ, R3f, 64, 16, 128, 128, 128, 16, false, tid);        // pq
  mm_f32(R4h, ws + OFF_PAMWK, R3f + 1024, 64, 16, 128, 128, 128, 16, false, tid); // pk
  mm_f16h(ws + OFF_PAMWV, R4h, pvh, 128, 64, 128, 128, 128, 64, pambv, tid);      // pv (c-major)
  __syncthreads();
  for (int idx = tid; idx < 49 * 16; idx += TPB) {
    R3f[idx]        += pambq[idx & 15];
    R3f[1024 + idx] += pambk[idx & 15];
  }
  __syncthreads();
  for (int n = tid; n < 64; n += TPB) {
    if (n < 49) {
      float sc[49];
      float mx = -1e30f;
      for (int m = 0; m < 49; ++m) {
        float s = 0.f;
        for (int c = 0; c < 16; ++c) s += R3f[n * 16 + c] * R3f[1024 + m * 16 + c];
        sc[m] = s;
        mx = fmaxf(mx, s);
      }
      float den = 0.f;
      for (int m = 0; m < 49; ++m) { sc[m] = __expf(sc[m] - mx); den += sc[m]; }
      const float inv = 1.f / den;
      for (int m = 0; m < 49; ++m) pattnh[n * 64 + m] = (_Float16)(sc[m] * inv);
      for (int m = 49; m < 64; ++m) pattnh[n * 64 + m] = (_Float16)0.f;
    } else {
      for (int m = 0; m < 64; ++m) pattnh[n * 64 + m] = (_Float16)0.f;
    }
  }
  __syncthreads();
  mm_f32(pattnh, pvh, R6f, 64, 128, 64, 64, 64, 128, false, tid); // op (n-major)
  __syncthreads();
  const float pamg = pamg_[0];
  for (int idx = tid; idx < 64 * 128; idx += TPB) {               // op' f16 -> R1
    const int n = idx >> 7;
    R1h[idx] = (n < 49) ? (_Float16)(pamg * R6f[idx] + R5f[idx]) : (_Float16)0.f;
  }
  __syncthreads();
  conv_block(R1h, ws + OFF_PAM, pambc, pambnw, pambnb, pamg, R6f, false, R5f, R7f, true, tid);
  __syncthreads();

  // ---- Phase F: MLP + pool + fc + layernorm ----
  for (int idx = tid; idx < 64 * 128; idx += TPB) {               // xm f16 -> R1
    const int n = idx >> 7;
    R1h[idx] = (n < 49) ? (_Float16)R7f[idx] : (_Float16)0.f;
  }
  __syncthreads();
  mm_gelu(R1h, ws + OFF_MLP1, hh, 64, 128, 128, 128, 128, 128, mlpb1, tid);
  __syncthreads();
  mm_f32(hh, ws + OFF_MLP2, R6f, 64, 128, 128, 128, 128, 128, false, tid);
  __syncthreads();
  for (int o = tid; o < 128; o += TPB) {
    float s = 0.f;
    for (int n = 0; n < 49; ++n) s += R6f[n * 128 + o];
    R3f[o] = s * (1.0f / 49.0f) + mlpb2[o];
  }
  __syncthreads();
  if (tid < 8) {
    float s = fcb[tid];
    for (int c = 0; c < 128; ++c) s += R3f[c] * fcw[tid * 128 + c];
    R3f[128 + tid] = s;
  }
  __syncthreads();
  if (tid < 8) {
    float mu = 0.f;
    for (int t = 0; t < 8; ++t) mu += R3f[128 + t];
    mu *= 0.125f;
    float var = 0.f;
    for (int t = 0; t < 8; ++t) { const float d = R3f[128 + t] - mu; var += d * d; }
    var *= 0.125f;
    const float nv = (R3f[128 + tid] - mu) * rsqrtf(var + 1e-5f);
    out[(size_t)b * 8 + tid] = nv * lnw[tid] + lnb[tid];
  }
}

// ---------------- host launcher ----------------
extern "C" void kernel_launch(void* const* d_in, const int* in_sizes, int n_in,
                              void* d_out, int out_size, void* d_ws, size_t ws_size,
                              hipStream_t stream) {
  (void)n_in; (void)out_size;
  if (ws_size < (size_t)WS_TOTAL * sizeof(_Float16)) return;  // workspace too small

  const float* xin = (const float*)d_in[0];
  _Float16* ws = (_Float16*)d_ws;
  const int B = in_sizes[0] / (14 * 49);

  const int pgrid = (WS_TOTAL + TPB - 1) / TPB;
  prep_weights<<<pgrid, TPB, 0, stream>>>(
      (const float*)d_in[9],  (const float*)d_in[10], (const float*)d_in[11],
      (const float*)d_in[12], (const float*)d_in[14], (const float*)d_in[15],
      (const float*)d_in[16], (const float*)d_in[17], (const float*)d_in[20],
      (const float*)d_in[31], (const float*)d_in[28], (const float*)d_in[24],
      (const float*)d_in[26], (const float*)d_in[35], (const float*)d_in[37], ws);

  fused_net<<<B, TPB, 163840, stream>>>(
      xin,
      (const float*)d_in[1],  (const float*)d_in[2],
      (const float*)d_in[3],  (const float*)d_in[4],
      (const float*)d_in[5],  (const float*)d_in[6],
      (const float*)d_in[7],  (const float*)d_in[8],
      (const float*)d_in[13], (const float*)d_in[18],
      (const float*)d_in[19], (const float*)d_in[21],
      (const float*)d_in[22], (const float*)d_in[23],
      (const float*)d_in[25], (const float*)d_in[27],
      (const float*)d_in[29], (const float*)d_in[30],
      (const float*)d_in[32], (const float*)d_in[33],
      (const float*)d_in[34],
      (const float*)d_in[36], (const float*)d_in[38],
      (const float*)d_in[39], (const float*)d_in[40],
      (const float*)d_in[41], (const float*)d_in[42],
      ws, (float*)d_out);
}
